// GRUCell_40003325395615
// MI455X (gfx1250) — matrix operand order repaired and checked by reading, not compiled, not run on hardware
//
#include <hip/hip_runtime.h>

// ---------------------------------------------------------------------------
// GRU cell fused kernel for MI455X (gfx1250, wave32, WMMA).
// B=16384, IN=H=512.  gi = x@Wi, gh = h@Wh (3 gates each), fused gate math.
// Strategy: bf16 WMMA (v_wmma_f32_16x16x32_bf16) with fp32 accumulation.
// Round 2: M-blocking (2 x 16-row subtiles per wave) so each weight (B)
// fragment feeds 2 WMMAs -> per-K-step mix 20 loads : 12 WMMAs (was 16 : 6).
// ---------------------------------------------------------------------------

typedef __bf16 bf16;
typedef bf16  v16bf __attribute__((ext_vector_type(16)));
typedef bf16  v8bf  __attribute__((ext_vector_type(8)));
typedef bf16  v4bf  __attribute__((ext_vector_type(4)));
typedef float v8f   __attribute__((ext_vector_type(8)));
typedef float v4f   __attribute__((ext_vector_type(4)));

#define B_ROWS 16384
#define KDIM   512
#define HDIM   512
#define WAVES  4   // waves per block (wave32 -> 128 threads)
#define MSUB   2   // 16-row M subtiles per wave -> M=32 rows per wave

// ---------------------------------------------------------------------------
// Prologue 1: fp32 -> bf16 conversion (row-major kept), 4 elems / thread.
// ---------------------------------------------------------------------------
__global__ void gru_cvt_bf16(const float* __restrict__ src,
                             bf16* __restrict__ dst, int n) {
    int i = (blockIdx.x * blockDim.x + threadIdx.x) * 4;
    if (i + 3 < n) {
        v4f v = *(const v4f*)(src + i);
        v4bf o;
        o[0] = (bf16)v[0]; o[1] = (bf16)v[1];
        o[2] = (bf16)v[2]; o[3] = (bf16)v[3];
        *(v4bf*)(dst + i) = o;
    }
}

// ---------------------------------------------------------------------------
// Prologue 2: transpose 512x512 fp32 W[k][n] -> bf16 WT[n][k] via LDS tile.
// WT layout lets a WMMA B-fragment be one contiguous 32B load per lane.
// ---------------------------------------------------------------------------
__global__ void gru_transpose_bf16(const float* __restrict__ src,
                                   bf16* __restrict__ dst) {
    __shared__ float tile[16][17];
    int k = blockIdx.y * 16 + threadIdx.y;   // source row (K)
    int n = blockIdx.x * 16 + threadIdx.x;   // source col (N)
    tile[threadIdx.y][threadIdx.x] = src[k * KDIM + n];
    __syncthreads();
    int on = blockIdx.x * 16 + threadIdx.y;  // dest row (N)
    int ok = blockIdx.y * 16 + threadIdx.x;  // dest col (K)
    dst[(size_t)on * KDIM + ok] = (bf16)tile[threadIdx.x][threadIdx.y];
}

// ---------------------------------------------------------------------------
// Fragment loaders using contiguous K order, the same for A and B (not the ISA table order).
// A 16x32 (MxK): lane holds row M=lane&15; elems 0..15 = 16 contiguous K
//                starting at kb + khalf, khalf = (lane>>4)*16.
// B 32x16 (KxN): lane holds col N=lane&15; elems 0..15 = 16 contiguous K
//                starting at kb + (lane>>4)*16  (from WT[n][k]).
// ---------------------------------------------------------------------------
__device__ __forceinline__ v16bf load_a_frag(const bf16* __restrict__ p) {
    union { v16bf v; struct { v8bf lo; v8bf hi; } s; } u;
    u.s.lo = *(const v8bf*)(p);        // K kb+khalf   .. +7
    u.s.hi = *(const v8bf*)(p + 8);    // K kb+8+khalf .. +7
    return u.v;
}
__device__ __forceinline__ v16bf load_b_frag(const bf16* __restrict__ p) {
    return *(const v16bf*)(p);         // 32 contiguous bytes (2x b128)
}

#define WMMA_BF16(a, b, c) \
    __builtin_amdgcn_wmma_f32_16x16x32_bf16(false, (a), false, (b), (short)0, (c), false, false)

__device__ __forceinline__ float sigmoid_f(float x) {
    return 1.0f / (1.0f + __expf(-x));
}

// ---------------------------------------------------------------------------
// Main fused kernel: each wave computes a 32(M)x16(N) tile of h_new as two
// 16x16 subtiles sharing the six weight (B) fragments.
// 12 accumulators; 16 K-steps x 12 WMMAs per wave.
// ---------------------------------------------------------------------------
__global__ __launch_bounds__(WAVES * 32, 1)
void gru_fused_wmma(const bf16* __restrict__ xb, const bf16* __restrict__ hb,
                    const bf16* __restrict__ wirT, const bf16* __restrict__ whrT,
                    const bf16* __restrict__ wizT, const bf16* __restrict__ whzT,
                    const bf16* __restrict__ winT, const bf16* __restrict__ whnT,
                    const float* __restrict__ b_r, const float* __restrict__ b_z,
                    const float* __restrict__ b_n,
                    const float* __restrict__ h_prev, float* __restrict__ out) {
    const int lane  = threadIdx.x & 31;
    const int wave  = threadIdx.x >> 5;
    const int tileM = (blockIdx.x * WAVES + wave) * (16 * MSUB);
    const int tileN = blockIdx.y * 16;

    const int lcol  = lane & 15;
    const int lhi   = lane >> 4;
    const int khalf = lhi * 16;

    // A-fragment row pointers: x and h, two 16-row subtiles each.
    const bf16* xrow0 = xb + (size_t)(tileM + lcol) * KDIM + khalf;
    const bf16* xrow1 = xrow0 + (size_t)16 * KDIM;
    const bf16* hrow0 = hb + (size_t)(tileM + lcol) * KDIM + khalf;
    const bf16* hrow1 = hrow0 + (size_t)16 * KDIM;

    // B-fragment pointers: six transposed weight matrices.
    const size_t boff = (size_t)(tileN + lcol) * KDIM + (size_t)lhi * 16;
    const bf16* pir = wirT + boff;
    const bf16* phr = whrT + boff;
    const bf16* piz = wizT + boff;
    const bf16* phz = whzT + boff;
    const bf16* pin = winT + boff;
    const bf16* phn = whnT + boff;

    v8f a_ir0 = {}, a_hr0 = {}, a_iz0 = {}, a_hz0 = {}, a_in0 = {}, a_hn0 = {};
    v8f a_ir1 = {}, a_hr1 = {}, a_iz1 = {}, a_hz1 = {}, a_in1 = {}, a_hn1 = {};

#pragma unroll 4
    for (int kb = 0; kb < KDIM; kb += 32) {
        // Prefetch the next K-slab of the activation streams (global_prefetch_b8).
        if (kb + 32 < KDIM) {
            __builtin_prefetch((const void*)(xrow0 + kb + 32), 0, 3);
            __builtin_prefetch((const void*)(xrow1 + kb + 32), 0, 3);
            __builtin_prefetch((const void*)(hrow0 + kb + 32), 0, 3);
            __builtin_prefetch((const void*)(hrow1 + kb + 32), 0, 3);
        }
        v16bf ax0 = load_a_frag(xrow0 + kb);
        v16bf ax1 = load_a_frag(xrow1 + kb);
        v16bf ah0 = load_a_frag(hrow0 + kb);
        v16bf ah1 = load_a_frag(hrow1 + kb);

        v16bf bir = load_b_frag(pir + kb);
        v16bf bhr = load_b_frag(phr + kb);
        v16bf biz = load_b_frag(piz + kb);
        v16bf bhz = load_b_frag(phz + kb);
        v16bf bin = load_b_frag(pin + kb);
        v16bf bhn = load_b_frag(phn + kb);

        // Each B fragment feeds two WMMAs (M subtile 0 and 1).
        a_ir0 = WMMA_BF16(ax0, bir, a_ir0);
        a_ir1 = WMMA_BF16(ax1, bir, a_ir1);
        a_hr0 = WMMA_BF16(ah0, bhr, a_hr0);
        a_hr1 = WMMA_BF16(ah1, bhr, a_hr1);
        a_iz0 = WMMA_BF16(ax0, biz, a_iz0);
        a_iz1 = WMMA_BF16(ax1, biz, a_iz1);
        a_hz0 = WMMA_BF16(ah0, bhz, a_hz0);
        a_hz1 = WMMA_BF16(ah1, bhz, a_hz1);
        a_in0 = WMMA_BF16(ax0, bin, a_in0);
        a_in1 = WMMA_BF16(ax1, bin, a_in1);
        a_hn0 = WMMA_BF16(ah0, bhn, a_hn0);
        a_hn1 = WMMA_BF16(ah1, bhn, a_hn1);
    }

    // Epilogue: C/D layout -> element (M = r + 8*lhi, N = lcol) per VGPR r.
    const int col = tileN + lcol;
    const float br = b_r[col];
    const float bz = b_z[col];
    const float bn = b_n[col];

#pragma unroll
    for (int s = 0; s < MSUB; ++s) {
        const v8f& cir = s ? a_ir1 : a_ir0;
        const v8f& chr = s ? a_hr1 : a_hr0;
        const v8f& ciz = s ? a_iz1 : a_iz0;
        const v8f& chz = s ? a_hz1 : a_hz0;
        const v8f& cin = s ? a_in1 : a_in0;
        const v8f& chn = s ? a_hn1 : a_hn0;
        const int mbase = tileM + s * 16 + lhi * 8;
#pragma unroll
        for (int r = 0; r < 8; ++r) {
            const int row = mbase + r;
            const size_t idx = (size_t)row * HDIM + col;
            float rt = sigmoid_f(cir[r] + chr[r] + br);
            float zt = sigmoid_f(ciz[r] + chz[r] + bz);
            float nt = tanhf(cin[r] + rt * chn[r] + bn);
            float hp = h_prev[idx];
            out[idx] = (1.0f - zt) * nt + zt * hp;
        }
    }
}

// ---------------------------------------------------------------------------
// Host-side launch.
// Inputs (setup_inputs order): 0 x_t, 1 h_t, 2 W_ir, 3 W_hr, 4 b_r,
//                              5 W_iz, 6 W_hz, 7 b_z, 8 W_in, 9 W_hn, 10 b_n
// Workspace: [x bf16 16MB][h bf16 16MB][6x WT bf16 3MB]  (~35 MB)
// ---------------------------------------------------------------------------
extern "C" void kernel_launch(void* const* d_in, const int* in_sizes, int n_in,
                              void* d_out, int out_size, void* d_ws, size_t ws_size,
                              hipStream_t stream) {
    const float* x    = (const float*)d_in[0];
    const float* h    = (const float*)d_in[1];
    const float* W_ir = (const float*)d_in[2];
    const float* W_hr = (const float*)d_in[3];
    const float* b_r  = (const float*)d_in[4];
    const float* W_iz = (const float*)d_in[5];
    const float* W_hz = (const float*)d_in[6];
    const float* b_z  = (const float*)d_in[7];
    const float* W_in = (const float*)d_in[8];
    const float* W_hn = (const float*)d_in[9];
    const float* b_n  = (const float*)d_in[10];
    float* out = (float*)d_out;

    char* ws = (char*)d_ws;
    const size_t act_bytes = (size_t)B_ROWS * KDIM * sizeof(bf16);
    const size_t w_elems   = (size_t)KDIM * HDIM;
    bf16* xb   = (bf16*)(ws);
    bf16* hb   = (bf16*)(ws + act_bytes);
    bf16* wt   = (bf16*)(ws + 2 * act_bytes);
    bf16* wirT = wt + 0 * w_elems;
    bf16* whrT = wt + 1 * w_elems;
    bf16* wizT = wt + 2 * w_elems;
    bf16* whzT = wt + 3 * w_elems;
    bf16* winT = wt + 4 * w_elems;
    bf16* whnT = wt + 5 * w_elems;

    // Prologue: convert activations (row-major bf16).
    const int nact = B_ROWS * KDIM;
    gru_cvt_bf16<<<nact / (256 * 4), 256, 0, stream>>>(x, xb, nact);
    gru_cvt_bf16<<<nact / (256 * 4), 256, 0, stream>>>(h, hb, nact);

    // Prologue: transpose+convert the six weight matrices.
    dim3 tb(16, 16), tg(HDIM / 16, KDIM / 16);
    gru_transpose_bf16<<<tg, tb, 0, stream>>>(W_ir, wirT);
    gru_transpose_bf16<<<tg, tb, 0, stream>>>(W_hr, whrT);
    gru_transpose_bf16<<<tg, tb, 0, stream>>>(W_iz, wizT);
    gru_transpose_bf16<<<tg, tb, 0, stream>>>(W_hz, whzT);
    gru_transpose_bf16<<<tg, tb, 0, stream>>>(W_in, winT);
    gru_transpose_bf16<<<tg, tb, 0, stream>>>(W_hn, whnT);

    // Main fused GEMM + gate kernel: block tile M=128 (4 waves x 32), N=16.
    dim3 gg(B_ROWS / (16 * MSUB * WAVES), HDIM / 16);
    gru_fused_wmma<<<gg, WAVES * 32, 0, stream>>>(
        xb, hb, wirT, whrT, wizT, whzT, winT, whnT,
        b_r, b_z, b_n, h, out);
}
